// OmegaMambaBlock_60533269070476
// MI455X (gfx1250) — compile-verified, matrix operand order checked by reading, not run on hardware
//
#include <hip/hip_runtime.h>
#include <math.h>

// ---------------- model constants ----------------
#define BATCH      2
#define SEQ        512
#define TTOK       (BATCH * SEQ)          // 1024 tokens
#define D_MODEL    2048
#define D_STATE    64
#define D_CONV     4
#define HEADDIM    64
#define D_INNER    4096
#define NHEADS     64
#define CONV_DIM   (D_INNER + 2 * D_STATE)              // 4224
#define D_IN_PROJ  (2 * D_INNER + 2 * D_STATE + NHEADS) // 8384
#define NUM_EXP    8
#define D_FF       8192
#define EPS        1e-6f

// ---------------- WMMA types ----------------
typedef __bf16 bf16_t;
typedef bf16_t v16bf __attribute__((ext_vector_type(16)));
typedef bf16_t v2bf  __attribute__((ext_vector_type(2)));
typedef float  v8f   __attribute__((ext_vector_type(8)));

union Frag { v16bf v; unsigned int u[8]; };

// f32 pair -> packed bf16x2 (RNE) in ONE VALU op.
// clang for gfx1250 does not expose the builtin, but CDNA keeps the
// v_cvt_pk_bf16_f32 instruction; emit it directly via inline asm.
__device__ __forceinline__ unsigned int pack_bf16x2(float lo, float hi) {
#if __has_builtin(__builtin_amdgcn_cvt_pk_bf16_f32)
  union { v2bf v; unsigned int u; } c;
  c.v = __builtin_amdgcn_cvt_pk_bf16_f32(lo, hi);
  return c.u;
#else
  unsigned int r;
  asm("v_cvt_pk_bf16_f32 %0, %1, %2" : "=v"(r) : "v"(lo), "v"(hi));
  return r;
#endif
}

// Load one 16x32 A/B fragment half for this lane: two 8-float chunks.
__device__ __forceinline__ void load_frag(const float* __restrict__ p0,
                                          const float* __restrict__ p1,
                                          Frag& f) {
  float4 x0 = *(const float4*)(p0);
  float4 x1 = *(const float4*)(p0 + 4);
  float4 y0 = *(const float4*)(p1);
  float4 y1 = *(const float4*)(p1 + 4);
  f.u[0] = pack_bf16x2(x0.x, x0.y); f.u[1] = pack_bf16x2(x0.z, x0.w);
  f.u[2] = pack_bf16x2(x1.x, x1.y); f.u[3] = pack_bf16x2(x1.z, x1.w);
  f.u[4] = pack_bf16x2(y0.x, y0.y); f.u[5] = pack_bf16x2(y0.z, y0.w);
  f.u[6] = pack_bf16x2(y1.x, y1.y); f.u[7] = pack_bf16x2(y1.z, y1.w);
}

__device__ __forceinline__ float silu_f(float v) {
  return v / (1.f + expf(-v));
}

#define WMMA_BF16(a, b, c) \
  __builtin_amdgcn_wmma_f32_16x16x32_bf16(false, (a), false, (b), (short)0, (c), false, false)

// =====================================================================
// bf16-WMMA GEMM:  C[M,N] = act(A[M,K] @ B[N,K]^T + bias) (+res)
// Wave tile 32x32 (4 WMMAs / K-step); block 256 thr = 8 waves = 64x128.
// Optional: row-gather on A, device M count, device row offsets on A/C,
// scatter-accumulate on C with per-row scale (MoE combine).
// Requires: N % 32 == 0, K % 32 == 0.
// =====================================================================
__global__ __launch_bounds__(256)
void omb_gemm_wmma_bf16(const float* __restrict__ A,
                        const float* __restrict__ Bw,
                        float* __restrict__ C,
                        const float* __restrict__ bias,      // [N] or null
                        const float* __restrict__ residual,  // [M,N] or null
                        const int*   __restrict__ gather,    // A row idx or null
                        const int*   __restrict__ scatter,   // C row idx or null
                        const float* __restrict__ scale,     // per-row C scale or null
                        const int*   __restrict__ dyn_m,     // device M or null
                        const int*   __restrict__ a_row_off, // device A row offset or null
                        const int*   __restrict__ c_row_off, // device C row offset or null
                        int M, int N, int K, int act)
{
  const int wave = threadIdx.x >> 5;
  const int lane = threadIdx.x & 31;
  const int m0 = blockIdx.y * 64 + (wave >> 2) * 32;  // 2 wave-rows
  const int n0 = blockIdx.x * 128 + (wave & 3) * 32;  // 4 wave-cols

  const int Meff = dyn_m ? *dyn_m : M;
  if (m0 >= Meff || n0 >= N) return;
  const int aoff = a_row_off ? *a_row_off : 0;
  const int coff = c_row_off ? *c_row_off : 0;

  const int half = lane >> 4;   // which K-chunk set this lane holds
  const int sub  = lane & 15;   // row (A) / col (B) within 16-tile

  int am0 = m0 + sub;       if (am0 >= Meff) am0 = Meff - 1;
  int am1 = m0 + 16 + sub;  if (am1 >= Meff) am1 = Meff - 1;
  const long ar0 = gather ? (long)gather[am0] : (long)(am0 + aoff);
  const long ar1 = gather ? (long)gather[am1] : (long)(am1 + aoff);
  const float* __restrict__ Ap0 = A + ar0 * (long)K + 8 * half;
  const float* __restrict__ Ap1 = A + ar1 * (long)K + 8 * half;
  const float* __restrict__ Bp0 = Bw + (long)(n0 + sub) * K + 8 * half;
  const float* __restrict__ Bp1 = Bw + (long)(n0 + 16 + sub) * K + 8 * half;

  v8f acc00 = {0.f,0.f,0.f,0.f,0.f,0.f,0.f,0.f};
  v8f acc01 = acc00, acc10 = acc00, acc11 = acc00;

#pragma unroll 2
  for (int kk = 0; kk < K; kk += 32) {
    Frag a0, a1, b0, b1;
    load_frag(Ap0 + kk, Ap0 + kk + 16, a0);
    load_frag(Ap1 + kk, Ap1 + kk + 16, a1);
    load_frag(Bp0 + kk, Bp0 + kk + 16, b0);
    load_frag(Bp1 + kk, Bp1 + kk + 16, b1);
    acc00 = WMMA_BF16(a0.v, b0.v, acc00);
    acc01 = WMMA_BF16(a0.v, b1.v, acc01);
    acc10 = WMMA_BF16(a1.v, b0.v, acc10);
    acc11 = WMMA_BF16(a1.v, b1.v, acc11);
  }

  // C/D layout: VGPR r -> M = 8*half + r ; lane%16 -> N
  v8f* accs[2][2] = { {&acc00, &acc01}, {&acc10, &acc11} };
#pragma unroll
  for (int tm = 0; tm < 2; ++tm) {
#pragma unroll
    for (int tn = 0; tn < 2; ++tn) {
      const int cn = n0 + tn * 16 + sub;
      const float bv = bias ? bias[cn] : 0.f;
      const v8f& acc = *accs[tm][tn];
#pragma unroll
      for (int r = 0; r < 8; ++r) {
        const int cm = m0 + tm * 16 + half * 8 + r;
        if (cm < Meff) {
          float v = acc[r] + bv;
          if (act == 1) v = silu_f(v);
          if (residual) v += residual[(long)cm * N + cn];
          if (scatter) {
            const float s = scale ? scale[cm] : 1.f;
            atomicAdd(&C[(long)scatter[cm] * N + cn], s * v);
          } else {
            C[((long)(cm + coff)) * N + cn] = v;
          }
        }
      }
    }
  }
}

// =====================================================================
// RMSNorm: one block per row
// =====================================================================
__global__ __launch_bounds__(256)
void omb_rmsnorm(const float* __restrict__ in, const float* __restrict__ w,
                 float* __restrict__ out, int Dw)
{
  const long row = blockIdx.x;
  const float* x = in + row * Dw;
  float* o = out + row * Dw;
  float ss = 0.f;
  for (int i = threadIdx.x; i < Dw; i += 256) { float v = x[i]; ss += v * v; }
  __shared__ float red[256];
  red[threadIdx.x] = ss; __syncthreads();
  for (int s = 128; s > 0; s >>= 1) {
    if (threadIdx.x < s) red[threadIdx.x] += red[threadIdx.x + s];
    __syncthreads();
  }
  const float sc = rsqrtf(red[0] / (float)Dw + EPS);
  for (int i = threadIdx.x; i < Dw; i += 256) o[i] = x[i] * sc * w[i];
}

// gated: out = rmsnorm(y * silu(z)) * w ; z has row stride ldz
__global__ __launch_bounds__(256)
void omb_gated_rmsnorm(const float* __restrict__ y, const float* __restrict__ z,
                       const float* __restrict__ w, float* __restrict__ out,
                       int Dw, int ldz)
{
  const long row = blockIdx.x;
  const float* yr = y + row * Dw;
  const float* zr = z + row * (long)ldz;
  float* o = out + row * Dw;
  float ss = 0.f;
  for (int i = threadIdx.x; i < Dw; i += 256) {
    float t = yr[i] * silu_f(zr[i]); ss += t * t;
  }
  __shared__ float red[256];
  red[threadIdx.x] = ss; __syncthreads();
  for (int s = 128; s > 0; s >>= 1) {
    if (threadIdx.x < s) red[threadIdx.x] += red[threadIdx.x + s];
    __syncthreads();
  }
  const float sc = rsqrtf(red[0] / (float)Dw + EPS);
  for (int i = threadIdx.x; i < Dw; i += 256)
    o[i] = yr[i] * silu_f(zr[i]) * sc * w[i];
}

// =====================================================================
// causal depthwise conv (width 4) + silu over xBC slice of zxbcdt
// =====================================================================
__global__ void omb_conv_silu(const float* __restrict__ zxbcdt,
                              const float* __restrict__ cw,
                              const float* __restrict__ cb,
                              float* __restrict__ xbc)
{
  const long idx = (long)blockIdx.x * blockDim.x + threadIdx.x;
  if (idx >= (long)TTOK * CONV_DIM) return;
  const int c = (int)(idx % CONV_DIM);
  const int t = (int)(idx / CONV_DIM);
  const int b = t / SEQ, l = t % SEQ;
  float acc = cb[c];
#pragma unroll
  for (int j = 0; j < D_CONV; ++j) {
    const int lj = l - (D_CONV - 1) + j;
    if (lj >= 0)
      acc += cw[c * D_CONV + j] *
             zxbcdt[((long)(b * SEQ + lj)) * D_IN_PROJ + D_INNER + c];
  }
  xbc[idx] = silu_f(acc);
}

// dt = softplus(raw + dt_bias); dA = exp(dt * (-exp(A_log)))
__global__ void omb_dt(const float* __restrict__ zxbcdt,
                       const float* __restrict__ dt_bias,
                       const float* __restrict__ A_log,
                       float* __restrict__ dt, float* __restrict__ dA)
{
  const int idx = blockIdx.x * blockDim.x + threadIdx.x;
  if (idx >= TTOK * NHEADS) return;
  const int h = idx % NHEADS, t = idx / NHEADS;
  const float r = zxbcdt[(long)t * D_IN_PROJ + D_INNER + CONV_DIM + h] + dt_bias[h];
  const float d = (r > 20.f) ? r : log1pf(expf(r));
  dt[idx] = d;
  dA[idx] = expf(d * (-expf(A_log[h])));
}

// =====================================================================
// SSM scan: one block per (batch, head); state 64x64 in registers
// =====================================================================
__global__ __launch_bounds__(256)
void omb_ssm_scan(const float* __restrict__ xbc,
                  const float* __restrict__ dt,
                  const float* __restrict__ dA,
                  const float* __restrict__ Dp,
                  float* __restrict__ y)
{
  const int bh = blockIdx.x;
  const int b = bh / NHEADS, h = bh % NHEADS;
  const int tid = threadIdx.x;
  const int p = tid & 63;          // headdim index
  const int ng = tid >> 6;         // 0..3 state-column group
  float st[16];
#pragma unroll
  for (int i = 0; i < 16; ++i) st[i] = 0.f;

  __shared__ float sx[64], sB[64], sC[64], part[256];
  const float dskip = Dp[h];

  for (int l = 0; l < SEQ; ++l) {
    const long t = (long)b * SEQ + l;
    __syncthreads();
    if (tid < 64)       sx[tid]       = xbc[t * CONV_DIM + h * HEADDIM + tid];
    else if (tid < 128) sB[tid - 64]  = xbc[t * CONV_DIM + D_INNER + (tid - 64)];
    else if (tid < 192) sC[tid - 128] = xbc[t * CONV_DIM + D_INNER + D_STATE + (tid - 128)];
    __syncthreads();
    const float dtv = dt[t * NHEADS + h];
    const float dAv = dA[t * NHEADS + h];
    const float xv = sx[p];
    float acc = 0.f;
#pragma unroll
    for (int i = 0; i < 16; ++i) {
      const int n = ng * 16 + i;
      const float hh = st[i] * dAv + dtv * xv * sB[n];
      st[i] = hh;
      acc += hh * sC[n];
    }
    part[tid] = acc;
    __syncthreads();
    if (tid < 64) {
      const float yv = part[tid] + part[tid + 64] + part[tid + 128] + part[tid + 192]
                     + dskip * sx[tid];
      y[t * D_INNER + h * HEADDIM + tid] = yv;
    }
  }
}

// =====================================================================
// Gate: logits, softmax probs, top-2 indices + renormalized weights
// =====================================================================
__global__ __launch_bounds__(256)
void omb_gate(const float* __restrict__ hn, const float* __restrict__ gw,
              float* __restrict__ logits, float* __restrict__ probs,
              int* __restrict__ idx_top, float* __restrict__ wt_top)
{
  const int t = blockIdx.x;
  float acc[NUM_EXP];
#pragma unroll
  for (int e = 0; e < NUM_EXP; ++e) acc[e] = 0.f;
  for (int k = threadIdx.x; k < D_MODEL; k += 256) {
    const float x = hn[(long)t * D_MODEL + k];
#pragma unroll
    for (int e = 0; e < NUM_EXP; ++e) acc[e] += x * gw[(long)e * D_MODEL + k];
  }
  __shared__ float red[NUM_EXP][256];
#pragma unroll
  for (int e = 0; e < NUM_EXP; ++e) red[e][threadIdx.x] = acc[e];
  __syncthreads();
  for (int s = 128; s > 0; s >>= 1) {
    if (threadIdx.x < s)
#pragma unroll
      for (int e = 0; e < NUM_EXP; ++e) red[e][threadIdx.x] += red[e][threadIdx.x + s];
    __syncthreads();
  }
  if (threadIdx.x == 0) {
    float lg[NUM_EXP], mx = -1e30f;
#pragma unroll
    for (int e = 0; e < NUM_EXP; ++e) { lg[e] = red[e][0]; logits[t * NUM_EXP + e] = lg[e]; mx = fmaxf(mx, lg[e]); }
    float den = 0.f;
#pragma unroll
    for (int e = 0; e < NUM_EXP; ++e) den += expf(lg[e] - mx);
#pragma unroll
    for (int e = 0; e < NUM_EXP; ++e) probs[t * NUM_EXP + e] = expf(lg[e] - mx) / den;
    int i0 = 0;
    for (int e = 1; e < NUM_EXP; ++e) if (lg[e] > lg[i0]) i0 = e;  // first max
    int i1 = (i0 == 0) ? 1 : 0;
    for (int e = 0; e < NUM_EXP; ++e) if (e != i0 && lg[e] > lg[i1]) i1 = e;
    const float e1 = expf(lg[i1] - lg[i0]);
    idx_top[t * 2] = i0; idx_top[t * 2 + 1] = i1;
    wt_top[t * 2] = 1.f / (1.f + e1);
    wt_top[t * 2 + 1] = e1 / (1.f + e1);
  }
}

// =====================================================================
// MoE compaction: deterministic (token-order) gather lists per expert
// =====================================================================
__global__ __launch_bounds__(1024)
void omb_moe_compact(const int* __restrict__ idx_top, const float* __restrict__ wt_top,
                     int* __restrict__ tok_list, float* __restrict__ tok_w,
                     int* __restrict__ counts, int* __restrict__ offsets)
{
  __shared__ int scan[TTOK];
  const int t = threadIdx.x;
  const int i0 = idx_top[t * 2], i1 = idx_top[t * 2 + 1];
  const float w0 = wt_top[t * 2], w1 = wt_top[t * 2 + 1];
  int running = 0;
  for (int e = 0; e < NUM_EXP; ++e) {
    const int flag = (i0 == e || i1 == e) ? 1 : 0;
    const float w = (i0 == e) ? w0 : ((i1 == e) ? w1 : 0.f);
    scan[t] = flag; __syncthreads();
    for (int s = 1; s < TTOK; s <<= 1) {
      const int v = scan[t];
      const int add = (t >= s) ? scan[t - s] : 0;
      __syncthreads();
      scan[t] = v + add;
      __syncthreads();
    }
    const int pos = scan[t] - flag;
    const int total = scan[TTOK - 1];
    if (flag) { tok_list[e * TTOK + pos] = t; tok_w[e * TTOK + pos] = w; }
    if (t == 0) { counts[e] = total; offsets[e] = running; }
    running += total;
    __syncthreads();
  }
}

// =====================================================================
// misc elementwise
// =====================================================================
__global__ void omb_zero(float* __restrict__ p, long n) {
  const long i = (long)blockIdx.x * blockDim.x + threadIdx.x;
  if (i < n) p[i] = 0.f;
}

__global__ void omb_add(const float* __restrict__ a, const float* __restrict__ b,
                        float* __restrict__ o, long n) {
  const long i = (long)blockIdx.x * blockDim.x + threadIdx.x;
  if (i < n) o[i] = a[i] + b[i];
}

// =====================================================================
// safety heads: one thread per token over f[T,128]
// =====================================================================
__global__ void omb_safety_heads(const float* __restrict__ f,
                                 const float* __restrict__ hw, const float* __restrict__ hb,
                                 const float* __restrict__ uw, const float* __restrict__ ub,
                                 const float* __restrict__ tw, const float* __restrict__ tb,
                                 float* __restrict__ risk_vals, float* __restrict__ risk_idx,
                                 float* __restrict__ unc, float* __restrict__ tox,
                                 float* __restrict__ harm)
{
  const int t = blockIdx.x * blockDim.x + threadIdx.x;
  if (t >= TTOK) return;
  const float* ft = f + (long)t * 128;
  float mx = -1.f; int mi = 0;
#pragma unroll
  for (int j = 0; j < 5; ++j) {
    float a = hb[j];
    for (int k = 0; k < 128; ++k) a += ft[k] * hw[j * 128 + k];
    harm[t * 5 + j] = a;
    const float s = 1.f / (1.f + expf(-a));
    if (s > mx) { mx = s; mi = j; }
  }
  risk_vals[t] = mx;
  risk_idx[t] = (float)mi;
  float au = ub[0], at = tb[0];
  for (int k = 0; k < 128; ++k) { au += ft[k] * uw[k]; at += ft[k] * tw[k]; }
  unc[t] = 1.f / (1.f + expf(-au));
  tox[t] = 1.f / (1.f + expf(-at));
}

// lb = E * sum_e mean_t(probs)^2
__global__ __launch_bounds__(256)
void omb_lb(const float* __restrict__ probs, float* __restrict__ out_lb)
{
  float acc[NUM_EXP];
#pragma unroll
  for (int e = 0; e < NUM_EXP; ++e) acc[e] = 0.f;
  for (int t = threadIdx.x; t < TTOK; t += 256)
#pragma unroll
    for (int e = 0; e < NUM_EXP; ++e) acc[e] += probs[t * NUM_EXP + e];
  __shared__ float red[NUM_EXP][256];
#pragma unroll
  for (int e = 0; e < NUM_EXP; ++e) red[e][threadIdx.x] = acc[e];
  __syncthreads();
  for (int s = 128; s > 0; s >>= 1) {
    if (threadIdx.x < s)
#pragma unroll
      for (int e = 0; e < NUM_EXP; ++e) red[e][threadIdx.x] += red[e][threadIdx.x + s];
    __syncthreads();
  }
  if (threadIdx.x == 0) {
    float lb = 0.f;
#pragma unroll
    for (int e = 0; e < NUM_EXP; ++e) {
      const float m = red[e][0] / (float)TTOK;
      lb += m * m;
    }
    out_lb[0] = (float)NUM_EXP * lb;
  }
}

// =====================================================================
// host orchestration
// =====================================================================
static inline void launch_gemm(hipStream_t s, const float* A, const float* B, float* C,
                               const float* bias, const float* res,
                               const int* gather, const int* scatter, const float* scale,
                               const int* dyn_m, const int* aoff, const int* coff,
                               int M, int N, int K, int act)
{
  dim3 grid((N + 127) / 128, (M + 63) / 64);
  omb_gemm_wmma_bf16<<<grid, 256, 0, s>>>(A, B, C, bias, res, gather, scatter, scale,
                                          dyn_m, aoff, coff, M, N, K, act);
}

extern "C" void kernel_launch(void* const* d_in, const int* in_sizes, int n_in,
                              void* d_out, int out_size, void* d_ws, size_t ws_size,
                              hipStream_t stream)
{
  const float* x          = (const float*)d_in[0];
  const float* in_proj_w  = (const float*)d_in[1];
  const float* conv_w     = (const float*)d_in[2];
  const float* conv_b     = (const float*)d_in[3];
  const float* dt_bias    = (const float*)d_in[4];
  const float* A_log      = (const float*)d_in[5];
  const float* Dp         = (const float*)d_in[6];
  const float* ssm_norm_w = (const float*)d_in[7];
  const float* out_proj_w = (const float*)d_in[8];
  const float* norm_mam_w = (const float*)d_in[9];
  const float* norm_moe_w = (const float*)d_in[10];
  const float* gate_w     = (const float*)d_in[11];
  const float* w1         = (const float*)d_in[12];
  const float* w2         = (const float*)d_in[13];
  const float* sp1_w      = (const float*)d_in[14];
  const float* sp1_b      = (const float*)d_in[15];
  const float* sp2_w      = (const float*)d_in[16];
  const float* sp2_b      = (const float*)d_in[17];
  const float* harm_w     = (const float*)d_in[18];
  const float* harm_b     = (const float*)d_in[19];
  const float* unc_w      = (const float*)d_in[20];
  const float* unc_b      = (const float*)d_in[21];
  const float* tox_w      = (const float*)d_in[22];
  const float* tox_b      = (const float*)d_in[23];

  // ---------------- workspace carve-up ----------------
  float* W = (float*)d_ws;
  size_t o = 0;
  auto alloc = [&](size_t n) { float* p = W + o; o += n; return p; };
  float* h_norm   = alloc((size_t)TTOK * D_MODEL);
  float* zxbcdt   = alloc((size_t)TTOK * D_IN_PROJ);
  float* xbc      = alloc((size_t)TTOK * CONV_DIM);
  float* dt       = alloc((size_t)TTOK * NHEADS);
  float* dA       = alloc((size_t)TTOK * NHEADS);
  float* y_ssm    = alloc((size_t)TTOK * D_INNER);
  float* y_gated  = alloc((size_t)TTOK * D_INNER);
  float* h_res    = alloc((size_t)TTOK * D_MODEL);
  float* hn       = alloc((size_t)TTOK * D_MODEL);
  float* logits   = alloc((size_t)TTOK * NUM_EXP);
  float* probs    = alloc((size_t)TTOK * NUM_EXP);
  float* wt_top   = alloc((size_t)TTOK * 2);
  float* tok_w    = alloc((size_t)NUM_EXP * TTOK);
  float* moe_out  = alloc((size_t)TTOK * D_MODEL);
  float* sf1      = alloc((size_t)TTOK * 1024);
  float* sf2      = alloc((size_t)TTOK * 128);
  float* h1moe    = alloc((size_t)2 * TTOK * D_FF);   // compacted, sum(counts)=2T
  int* tok_list = (int*)(W + o); o += (size_t)NUM_EXP * TTOK;
  int* idx_top  = (int*)(W + o); o += (size_t)TTOK * 2;
  int* counts   = (int*)(W + o); o += NUM_EXP;
  int* offsets  = (int*)(W + o); o += NUM_EXP;
  (void)ws_size; (void)in_sizes; (void)n_in; (void)out_size;

  // ---------------- output carve-up ----------------
  float* out_h    = (float*)d_out;                          // T*D_MODEL
  float* out_rval = out_h + (size_t)TTOK * D_MODEL;         // T
  float* out_ridx = out_rval + TTOK;                        // T
  float* out_unc  = out_ridx + TTOK;                        // T
  float* out_tox  = out_unc + TTOK;                         // T
  float* out_harm = out_tox + TTOK;                         // T*5
  float* out_lb   = out_harm + (size_t)TTOK * 5;            // 1

  // ---------------- Mamba2 ----------------
  omb_rmsnorm<<<TTOK, 256, 0, stream>>>(x, norm_mam_w, h_norm, D_MODEL);
  launch_gemm(stream, h_norm, in_proj_w, zxbcdt, nullptr, nullptr,
              nullptr, nullptr, nullptr, nullptr, nullptr, nullptr,
              TTOK, D_IN_PROJ, D_MODEL, 0);
  {
    long n = (long)TTOK * CONV_DIM;
    omb_conv_silu<<<(int)((n + 255) / 256), 256, 0, stream>>>(zxbcdt, conv_w, conv_b, xbc);
  }
  omb_dt<<<(TTOK * NHEADS + 255) / 256, 256, 0, stream>>>(zxbcdt, dt_bias, A_log, dt, dA);
  omb_ssm_scan<<<BATCH * NHEADS, 256, 0, stream>>>(xbc, dt, dA, Dp, y_ssm);
  omb_gated_rmsnorm<<<TTOK, 256, 0, stream>>>(y_ssm, zxbcdt /* z = cols [0,D_INNER) */,
                                              ssm_norm_w, y_gated, D_INNER, D_IN_PROJ);
  launch_gemm(stream, y_gated, out_proj_w, h_res, nullptr, x /* +residual */,
              nullptr, nullptr, nullptr, nullptr, nullptr, nullptr,
              TTOK, D_MODEL, D_INNER, 0);

  // ---------------- MoE (top-2 sparse, math-equivalent to dense dispatch) ----------------
  omb_rmsnorm<<<TTOK, 256, 0, stream>>>(h_res, norm_moe_w, hn, D_MODEL);
  omb_gate<<<TTOK, 256, 0, stream>>>(hn, gate_w, logits, probs, idx_top, wt_top);
  omb_moe_compact<<<1, TTOK, 0, stream>>>(idx_top, wt_top, tok_list, tok_w, counts, offsets);
  {
    long n = (long)TTOK * D_MODEL;
    omb_zero<<<(int)((n + 255) / 256), 256, 0, stream>>>(moe_out, n);
  }
  for (int e = 0; e < NUM_EXP; ++e) {
    // h1 = silu(gather(hn) @ w1[e]^T), written at device row offset offsets[e]
    launch_gemm(stream, hn, w1 + (size_t)e * D_FF * D_MODEL, h1moe,
                nullptr, nullptr, tok_list + e * TTOK, nullptr, nullptr,
                counts + e, nullptr, offsets + e,
                TTOK, D_FF, D_MODEL, 1);
    // moe_out[tok] += wt * (h1 @ w2[e]^T)
    launch_gemm(stream, h1moe, w2 + (size_t)e * D_MODEL * D_FF, moe_out,
                nullptr, nullptr, nullptr, tok_list + e * TTOK, tok_w + e * TTOK,
                counts + e, offsets + e, nullptr,
                TTOK, D_MODEL, D_FF, 0);
  }
  {
    long n = (long)TTOK * D_MODEL;
    omb_add<<<(int)((n + 255) / 256), 256, 0, stream>>>(h_res, moe_out, out_h, n);
  }

  // ---------------- safety ----------------
  launch_gemm(stream, out_h, sp1_w, sf1, sp1_b, nullptr,
              nullptr, nullptr, nullptr, nullptr, nullptr, nullptr,
              TTOK, 1024, D_MODEL, 1);
  launch_gemm(stream, sf1, sp2_w, sf2, sp2_b, nullptr,
              nullptr, nullptr, nullptr, nullptr, nullptr, nullptr,
              TTOK, 128, 1024, 0);
  omb_safety_heads<<<(TTOK + 127) / 128, 128, 0, stream>>>(
      sf2, harm_w, harm_b, unc_w, unc_b, tox_w, tox_b,
      out_rval, out_ridx, out_unc, out_tox, out_harm);
  omb_lb<<<1, 256, 0, stream>>>(probs, out_lb);
}
